// TransformerBlock_56075093016585
// MI455X (gfx1250) — matrix operand order repaired and checked by reading, not compiled, not run on hardware
//
#include <hip/hip_runtime.h>
#include <hip/hip_bf16.h>

#define EMBED   1024
#define HEADS   16
#define HEAD_DIM 64
#define HIDDEN  4096
#define BATCH   2
#define SEQ     2048
#define TOKENS  (BATCH * SEQ)   // 4096

typedef __attribute__((ext_vector_type(16))) __bf16 v16bf;
typedef __attribute__((ext_vector_type(8)))  __bf16 v8bf;
typedef __attribute__((ext_vector_type(2)))  __bf16 v2bf;
typedef __attribute__((ext_vector_type(8)))  float  v8f;

// Native bf16 conversion: backend picks hardware cvt (v_cvt_pk_bf16_f32).
static __device__ __forceinline__ unsigned pack2(float a, float b) {
    v2bf v; v[0] = (__bf16)a; v[1] = (__bf16)b;
    return __builtin_bit_cast(unsigned, v);
}
static __device__ __forceinline__ unsigned short f2bf(float f) {
    __bf16 h = (__bf16)f;
    return __builtin_bit_cast(unsigned short, h);
}
static __device__ __forceinline__ v8bf ld8(const unsigned short* p) {
    return *(const v8bf*)p;   // 16B aligned by construction -> ds_load_b128
}
static __device__ __forceinline__ v16bf join16(v8bf lo, v8bf hi) {
    return __builtin_shufflevector(lo, hi, 0,1,2,3,4,5,6,7,8,9,10,11,12,13,14,15);
}
static __device__ __forceinline__ v8f vzero8() {
    v8f r;
#pragma unroll
    for (int i = 0; i < 8; ++i) r[i] = 0.f;
    return r;
}

// ---------------------------------------------------------------------------
// Tiled GEMM: C[M,N] = A[M,K] * W[K,N] + bias, optional ReLU, optional bf16 io.
// Block 256 threads (8 waves), tile 128x128, BK=32.
// Double-buffered LDS with register prefetch: loads for tile k+1 issued
// before the WMMAs of tile k; one barrier per K-step.
// ---------------------------------------------------------------------------
template <bool A_BF16, bool OUT_BF16, bool RELU>
__global__ __launch_bounds__(256) void gemm_kernel(
    const void* __restrict__ Av, const float* __restrict__ W,
    const float* __restrict__ bias, void* __restrict__ Cv,
    int Kdim, int Ndim)
{
    __shared__ __align__(16) unsigned short As[2][128 * 40];
    __shared__ __align__(16) unsigned short Bs[2][128 * 40];   // W^T: [n][k]

    const int t    = threadIdx.x;
    const int lane = t & 31;
    const int wv   = t >> 5;
    const int half = lane >> 4;
    const int l15  = lane & 15;
    const int m0   = blockIdx.y * 128;
    const int n0   = blockIdx.x * 128;
    const int rBase = (wv >> 1) * 32;
    const int cBase = (wv & 1) * 64;

    const int aR = t >> 3, aC = t & 7;            // A: 32 rows/pass, 8 thr/row
    const int wK = (t >> 4) << 1, wC = t & 15;    // W: k-pair, 16B col chunk

    float4 aRegF[4];
    uint2  aRegU[4];
    float4 wReg[2][2];

    auto issueLoads = [&](int k0) {
#pragma unroll
        for (int p = 0; p < 4; ++p) {
            const int row = p * 32 + aR;
            if (A_BF16)
                aRegU[p] = *(const uint2*)((const unsigned short*)Av +
                              (size_t)(m0 + row) * Kdim + k0 + aC * 4);
            else
                aRegF[p] = *(const float4*)((const float*)Av +
                              (size_t)(m0 + row) * Kdim + k0 + aC * 4);
        }
#pragma unroll
        for (int p = 0; p < 2; ++p) {
            const int n = (wC + p * 16) * 4;
            wReg[p][0] = *(const float4*)(W + (size_t)(k0 + wK    ) * Ndim + n0 + n);
            wReg[p][1] = *(const float4*)(W + (size_t)(k0 + wK + 1) * Ndim + n0 + n);
        }
    };
    auto stageLDS = [&](int buf) {
#pragma unroll
        for (int p = 0; p < 4; ++p) {
            const int row = p * 32 + aR;
            uint2 d;
            if (A_BF16) d = aRegU[p];
            else { d.x = pack2(aRegF[p].x, aRegF[p].y); d.y = pack2(aRegF[p].z, aRegF[p].w); }
            *(uint2*)&As[buf][row * 40 + aC * 4] = d;
        }
#pragma unroll
        for (int p = 0; p < 2; ++p) {
            const int n = (wC + p * 16) * 4;
            const float* lo = (const float*)&wReg[p][0];
            const float* hi = (const float*)&wReg[p][1];
#pragma unroll
            for (int j = 0; j < 4; ++j)   // packed (k,k+1) pair -> b32 store
                *(unsigned*)&Bs[buf][(n + j) * 40 + wK] = pack2(lo[j], hi[j]);
        }
    };

    v8f acc[2][4];
#pragma unroll
    for (int mt = 0; mt < 2; ++mt)
#pragma unroll
        for (int nt = 0; nt < 4; ++nt) acc[mt][nt] = vzero8();

    const int nsteps = Kdim >> 5;
    issueLoads(0);
    stageLDS(0);
    __syncthreads();

    for (int kt = 0; kt < nsteps; ++kt) {
        const int buf = kt & 1;
        if (kt + 1 < nsteps) issueLoads((kt + 1) << 5);

        const unsigned short* Ab = As[buf];
        const unsigned short* Bb = Bs[buf];
        v16bf af[2], bf[4];
#pragma unroll
        for (int mt = 0; mt < 2; ++mt) {
            const int m = rBase + mt * 16 + l15;
            af[mt] = join16(ld8(&Ab[m * 40 + half * 8]),
                            ld8(&Ab[m * 40 + 16 + half * 8]));
        }
#pragma unroll
        for (int nt = 0; nt < 4; ++nt) {
            const int n = cBase + nt * 16 + l15;
            bf[nt] = join16(ld8(&Bb[n * 40 + half * 8]),
                            ld8(&Bb[n * 40 + 16 + half * 8]));
        }
#pragma unroll
        for (int mt = 0; mt < 2; ++mt)
#pragma unroll
            for (int nt = 0; nt < 4; ++nt)
                acc[mt][nt] = __builtin_amdgcn_wmma_f32_16x16x32_bf16(
                    false, af[mt], false, bf[nt], (short)0, acc[mt][nt], false, false);

        if (kt + 1 < nsteps) stageLDS(buf ^ 1);
        __syncthreads();
    }

    float bv[4];
#pragma unroll
    for (int nt = 0; nt < 4; ++nt) bv[nt] = bias[n0 + cBase + nt * 16 + l15];
#pragma unroll
    for (int mt = 0; mt < 2; ++mt)
#pragma unroll
        for (int nt = 0; nt < 4; ++nt)
#pragma unroll
            for (int d = 0; d < 8; ++d) {
                float v = acc[mt][nt][d] + bv[nt];
                if (RELU) v = fmaxf(v, 0.f);
                const int row = m0 + rBase + mt * 16 + half * 8 + d;
                const int col = n0 + cBase + nt * 16 + l15;
                if (OUT_BF16)
                    ((unsigned short*)Cv)[(size_t)row * Ndim + col] = f2bf(v);
                else
                    ((float*)Cv)[(size_t)row * Ndim + col] = v;
            }
}

// ---------------------------------------------------------------------------
// Flash attention, transposed-score formulation.
// One block = (128 queries, 1 head, 1 batch), 8 waves; wave owns 16 queries.
//   S^T = K * Q^T      (A = K block rows, B = Q^T)  -> lane holds one q column
//   softmax: in-lane reduction over 32 regs + one xor-16 shuffle; scalar stats
//   O^T = V^T * P^T    (A = V^T, B = P^T from packed-b128 LDS round trip)
// 1/sqrt(d) folded into Q staging. K/V double-buffered with register prefetch.
// ---------------------------------------------------------------------------
__global__ __launch_bounds__(256) void attn_kernel(
    const float* __restrict__ Q, const float* __restrict__ Kg,
    const float* __restrict__ Vg, float* __restrict__ ctx)
{
    __shared__ __align__(16) unsigned short Qs[128 * 72];    // [q][d]   (B^T for S^T)
    __shared__ __align__(16) unsigned short Ks[2][64 * 72];  // [key][d] (A for S^T)
    __shared__ __align__(16) unsigned short Vs[2][64 * 72];  // [d][key] (A for O^T)
    __shared__ __align__(16) unsigned short Ps[128 * 72];    // [q][key] (B^T for O^T)

    const int t    = threadIdx.x;
    const int lane = t & 31;
    const int wv   = t >> 5;
    const int half = lane >> 4;
    const int l15  = lane & 15;
    const int q0   = blockIdx.x * 128;
    const int h    = blockIdx.y;
    const int b    = blockIdx.z;
    const size_t base = (size_t)b * SEQ * EMBED + (size_t)h * HEAD_DIM;

    const int sR = t >> 4, sC = t & 15;   // staging: 16 rows/pass, 16 thr/row

    // ---- stage Q block (128x64) as bf16, pre-scaled by 1/sqrt(64) ----
#pragma unroll
    for (int p = 0; p < 8; ++p) {
        const int row = p * 16 + sR;
        float4 f = *(const float4*)(Q + base + (size_t)(q0 + row) * EMBED + sC * 4);
        uint2 d;
        d.x = pack2(f.x * 0.125f, f.y * 0.125f);
        d.y = pack2(f.z * 0.125f, f.w * 0.125f);
        *(uint2*)&Qs[row * 72 + sC * 4] = d;
    }

    // ---- K/V register prefetch + LDS staging ----
    float4 kReg[4];
    float4 vReg[2][2];
    auto issueKV = [&](int kb) {
#pragma unroll
        for (int p = 0; p < 4; ++p) {
            const int row = p * 16 + sR;
            kReg[p] = *(const float4*)(Kg + base + (size_t)(kb * 64 + row) * EMBED + sC * 4);
        }
#pragma unroll
        for (int p = 0; p < 2; ++p) {
            const int row = ((t >> 4) + p * 16) * 2;   // key pair
            vReg[p][0] = *(const float4*)(Vg + base + (size_t)(kb * 64 + row    ) * EMBED + sC * 4);
            vReg[p][1] = *(const float4*)(Vg + base + (size_t)(kb * 64 + row + 1) * EMBED + sC * 4);
        }
    };
    auto stageKV = [&](int buf) {
#pragma unroll
        for (int p = 0; p < 4; ++p) {
            const int row = p * 16 + sR;
            uint2 d; d.x = pack2(kReg[p].x, kReg[p].y); d.y = pack2(kReg[p].z, kReg[p].w);
            *(uint2*)&Ks[buf][row * 72 + sC * 4] = d;
        }
#pragma unroll
        for (int p = 0; p < 2; ++p) {
            const int row = ((t >> 4) + p * 16) * 2;
            const float* lo = (const float*)&vReg[p][0];
            const float* hi = (const float*)&vReg[p][1];
#pragma unroll
            for (int j = 0; j < 4; ++j)   // packed (key,key+1) -> b32 store
                *(unsigned*)&Vs[buf][(sC * 4 + j) * 72 + row] = pack2(lo[j], hi[j]);
        }
    };

    issueKV(0);
    stageKV(0);
    __syncthreads();

    // ---- Q^T B-fragments, register resident for the whole kernel ----
    const int prow = wv * 16 + l15;   // this lane's query row (both halves same q)
    v16bf qb[2];
#pragma unroll
    for (int ks = 0; ks < 2; ++ks)
        qb[ks] = join16(ld8(&Qs[prow * 72 + ks * 32 + half * 8]),
                        ld8(&Qs[prow * 72 + ks * 32 + 16 + half * 8]));

    float mr = -1e30f, lr = 0.f;
    v8f o[4];
#pragma unroll
    for (int dt = 0; dt < 4; ++dt) o[dt] = vzero8();

    const int nkb = SEQ / 64;
    for (int kb = 0; kb < nkb; ++kb) {
        const int buf = kb & 1;
        if (kb + 1 < nkb) issueKV(kb + 1);   // prefetch next key block

        // ---- S^T = K Q^T : 64 keys x 16 q per wave ----
        v8f s[4];
#pragma unroll
        for (int kt = 0; kt < 4; ++kt) s[kt] = vzero8();
#pragma unroll
        for (int ks = 0; ks < 2; ++ks)
#pragma unroll
            for (int kt = 0; kt < 4; ++kt) {
                const int mk = kt * 16 + l15;
                v16bf kf = join16(ld8(&Ks[buf][mk * 72 + ks * 32 + half * 8]),
                                  ld8(&Ks[buf][mk * 72 + ks * 32 + 16 + half * 8]));
                s[kt] = __builtin_amdgcn_wmma_f32_16x16x32_bf16(
                    false, kf, false, qb[ks], (short)0, s[kt], false, false);
            }

        // ---- online softmax: in-lane over 32 regs + one cross-half shuffle ----
        float mx = s[0][0];
#pragma unroll
        for (int kt = 0; kt < 4; ++kt)
#pragma unroll
            for (int d = 0; d < 8; ++d) mx = fmaxf(mx, s[kt][d]);
        mx = fmaxf(mx, __shfl_xor(mx, 16, 32));
        const float mnew  = fmaxf(mr, mx);
        const float alpha = __expf(mr - mnew);
        float sum = 0.f;
#pragma unroll
        for (int kt = 0; kt < 4; ++kt)
#pragma unroll
            for (int d = 0; d < 8; ++d) {
                float pj = __expf(s[kt][d] - mnew);
                s[kt][d] = pj;
                sum += pj;
            }
        sum += __shfl_xor(sum, 16, 32);
        lr = lr * alpha + sum;
        mr = mnew;
#pragma unroll
        for (int dt = 0; dt < 4; ++dt)
#pragma unroll
            for (int d = 0; d < 8; ++d) o[dt][d] *= alpha;

        // ---- P^T -> Ps[q][key]: one packed b128 store per key tile ----
#pragma unroll
        for (int kt = 0; kt < 4; ++kt) {
            uint4 pk;
            pk.x = pack2(s[kt][0], s[kt][1]);
            pk.y = pack2(s[kt][2], s[kt][3]);
            pk.z = pack2(s[kt][4], s[kt][5]);
            pk.w = pack2(s[kt][6], s[kt][7]);
            *(uint4*)&Ps[prow * 72 + kt * 16 + half * 8] = pk;
        }

        // ---- O^T += V^T P^T ----
#pragma unroll
        for (int ks = 0; ks < 2; ++ks) {
            v16bf pb = join16(ld8(&Ps[prow * 72 + ks * 32 + half * 8]),
                              ld8(&Ps[prow * 72 + ks * 32 + 16 + half * 8]));
#pragma unroll
            for (int dt = 0; dt < 4; ++dt) {
                const int md = dt * 16 + l15;
                v16bf va = join16(ld8(&Vs[buf][md * 72 + ks * 32 + half * 8]),
                                  ld8(&Vs[buf][md * 72 + ks * 32 + 16 + half * 8]));
                o[dt] = __builtin_amdgcn_wmma_f32_16x16x32_bf16(
                    false, va, false, pb, (short)0, o[dt], false, false);
            }
        }

        if (kb + 1 < nkb) stageKV(buf ^ 1);
        __syncthreads();
    }

    // ---- finalize + store O^T (lane owns query row, 8 consecutive d) ----
    const float inv = 1.f / lr;
    float* dst = ctx + base + (size_t)(q0 + prow) * EMBED;
#pragma unroll
    for (int dt = 0; dt < 4; ++dt) {
        float4 f0, f1;
        f0.x = o[dt][0] * inv; f0.y = o[dt][1] * inv;
        f0.z = o[dt][2] * inv; f0.w = o[dt][3] * inv;
        f1.x = o[dt][4] * inv; f1.y = o[dt][5] * inv;
        f1.z = o[dt][6] * inv; f1.w = o[dt][7] * inv;
        *(float4*)(dst + dt * 16 + half * 8)     = f0;
        *(float4*)(dst + dt * 16 + half * 8 + 4) = f1;
    }
}

// ---------------------------------------------------------------------------
// Residual + LayerNorm: out = LN(a + r) * g + b.  One block per token.
// ---------------------------------------------------------------------------
__global__ __launch_bounds__(256) void ln_kernel(
    const float* __restrict__ a, const float* __restrict__ r,
    const float* __restrict__ g, const float* __restrict__ be,
    float* __restrict__ out)
{
    __shared__ float r1[256], r2[256];
    __shared__ float mu_s, rs_s;
    const int tok = blockIdx.x;
    const int t   = threadIdx.x;
    const size_t base = (size_t)tok * EMBED;

    float v[4], s = 0.f, sq = 0.f;
#pragma unroll
    for (int i = 0; i < 4; ++i) {
        const int idx = t + i * 256;
        v[i] = a[base + idx] + r[base + idx];
        s  += v[i];
        sq += v[i] * v[i];
    }
    r1[t] = s; r2[t] = sq;
    __syncthreads();
    for (int off = 128; off > 0; off >>= 1) {
        if (t < off) { r1[t] += r1[t + off]; r2[t] += r2[t + off]; }
        __syncthreads();
    }
    if (t == 0) {
        float mu  = r1[0] * (1.f / EMBED);
        float var = r2[0] * (1.f / EMBED) - mu * mu;
        mu_s = mu;
        rs_s = rsqrtf(var + 1e-6f);
    }
    __syncthreads();
#pragma unroll
    for (int i = 0; i < 4; ++i) {
        const int idx = t + i * 256;
        out[base + idx] = (v[i] - mu_s) * rs_s * g[idx] + be[idx];
    }
}

// ---------------------------------------------------------------------------
extern "C" void kernel_launch(void* const* d_in, const int* in_sizes, int n_in,
                              void* d_out, int out_size, void* d_ws, size_t ws_size,
                              hipStream_t stream)
{
    (void)in_sizes; (void)n_in; (void)out_size; (void)ws_size;
    const float* x   = (const float*)d_in[0];
    const float* Wq  = (const float*)d_in[1];
    const float* bq  = (const float*)d_in[2];
    const float* Wk  = (const float*)d_in[3];
    const float* bk  = (const float*)d_in[4];
    const float* Wv  = (const float*)d_in[5];
    const float* bv  = (const float*)d_in[6];
    const float* Wo  = (const float*)d_in[7];
    const float* bo  = (const float*)d_in[8];
    const float* g1  = (const float*)d_in[9];
    const float* be1 = (const float*)d_in[10];
    const float* W1  = (const float*)d_in[11];
    const float* b1  = (const float*)d_in[12];
    const float* W2  = (const float*)d_in[13];
    const float* b2  = (const float*)d_in[14];
    const float* g2  = (const float*)d_in[15];
    const float* be2 = (const float*)d_in[16];

    const size_t SZ = (size_t)TOKENS * EMBED;   // 4096*1024
    float* ws   = (float*)d_ws;
    float* Qb   = ws;
    float* Kb   = ws + SZ;
    float* Vb   = ws + 2 * SZ;
    float* ctx  = ws + 3 * SZ;
    float* x1   = ws + 4 * SZ;
    float* attn = Vb;                              // reuse (V consumed)
    float* ffn  = ctx;                             // reuse (ctx consumed)
    unsigned short* hbf = (unsigned short*)ws;     // bf16 hidden, reuses Q+K regions

    dim3 blk(256);
    dim3 g_e(EMBED / 128, TOKENS / 128);   // (8, 32)
    dim3 g_h(HIDDEN / 128, TOKENS / 128);  // (32, 32)

    gemm_kernel<false, false, false><<<g_e, blk, 0, stream>>>(x, Wq, bq, Qb, EMBED, EMBED);
    gemm_kernel<false, false, false><<<g_e, blk, 0, stream>>>(x, Wk, bk, Kb, EMBED, EMBED);
    gemm_kernel<false, false, false><<<g_e, blk, 0, stream>>>(x, Wv, bv, Vb, EMBED, EMBED);

    attn_kernel<<<dim3(SEQ / 128, HEADS, BATCH), dim3(256), 0, stream>>>(Qb, Kb, Vb, ctx);

    gemm_kernel<false, false, false><<<g_e, blk, 0, stream>>>(ctx, Wo, bo, attn, EMBED, EMBED);
    ln_kernel<<<dim3(TOKENS), dim3(256), 0, stream>>>(x, attn, g1, be1, x1);

    gemm_kernel<false, true, true><<<g_h, blk, 0, stream>>>(x1, W1, b1, hbf, EMBED, HIDDEN);
    gemm_kernel<true, false, false><<<g_e, blk, 0, stream>>>(hbf, W2, b2, ffn, HIDDEN, EMBED);

    ln_kernel<<<dim3(TOKENS), dim3(256), 0, stream>>>(x1, ffn, g2, be2, (float*)d_out);
}
